// SwinTransformer_86260123172919
// MI455X (gfx1250) — compile-verified, matrix operand order checked by reading, not run on hardware
//
#include <hip/hip_runtime.h>
#include <hip/hip_bf16.h>

// ---------------- types / helpers ----------------
typedef _Float16 h16;
typedef __attribute__((ext_vector_type(16))) _Float16 v16h;
typedef __attribute__((ext_vector_type(8)))  _Float16 v8h;
typedef __attribute__((ext_vector_type(4)))  _Float16 v4h;
typedef __attribute__((ext_vector_type(8)))  float    v8f;
typedef __attribute__((ext_vector_type(4)))  float    v4f;

#define DEVI __device__ __forceinline__

// Problem constants
#define BATCH 32
#define HP    64
#define CC    128
#define WS_   8
#define SHIFT_ 4
#define NH    4
#define SS    64            // tokens per window
#define NWIN  64            // windows per image
#define HD    32            // head dim
#define NWT   2048          // total windows (BATCH*NWIN)
#define TOK   131072        // total tokens
#define NMLP  512
#define QSCALE 0.17677669529663687f

DEVI int lane_id() { return (int)(threadIdx.x & 31); }

// Load a 16x32 f16 A-fragment (or B-fragment of a transposed matrix) from a
// row-major matrix. Per ISA 7.12.2: lane L (0..15) holds row L, K = {0..7,16..23};
// lane L+16 holds row L, K = {8..15,24..31}. elems 0..7 -> first chunk, 8..15 -> second.
DEVI v16h load_frag(const h16* base, int ld) {
  int l = lane_id();
  int r = l & 15;
  int hi = l >> 4;                 // 0 or 1
  const h16* p = base + r * ld + hi * 8;
  v8h a = *(const v8h*)(p);
  v8h b = *(const v8h*)(p + 16);
  v16h out;
#pragma unroll
  for (int i = 0; i < 8; ++i) { out[i] = a[i]; out[8 + i] = b[i]; }
  return out;
}

DEVI v8f wmma_f16(v16h a, v16h b, v8f c) {
  return __builtin_amdgcn_wmma_f32_16x16x32_f16(false, a, false, b, (short)0, c, false, false);
}

// Two 16x16 output tiles (ni0, ni1) of A(16x128) @ B^T(n-major, ld=128), K=128.
// All 8 B-fragments are preloaded (clausable), then two independent 4-deep
// wmma accumulator chains are interleaved for ILP.
DEVI void gemm4x2(const v16h* a, const h16* bT, int ni0, int ni1, v8f* o0, v8f* o1) {
  __builtin_prefetch(bT + (ni0 + 4) * 16 * CC, 0, 3);
  v16h b0[4], b1[4];
#pragma unroll
  for (int ks = 0; ks < 4; ++ks) {
    b0[ks] = load_frag(bT + ni0 * 16 * CC + ks * 32, CC);
    b1[ks] = load_frag(bT + ni1 * 16 * CC + ks * 32, CC);
  }
  v8f A0 = {}, A1 = {};
#pragma unroll
  for (int ks = 0; ks < 4; ++ks) {
    A0 = wmma_f16(a[ks], b0[ks], A0);
    A1 = wmma_f16(a[ks], b1[ks], A1);
  }
  *o0 = A0;
  *o1 = A1;
}

// ---------------- K0: weight prep (f32 -> f16, transposed) ----------------
__global__ void k_prep_w(const float* __restrict__ qkv_w, const float* __restrict__ proj_w,
                         const float* __restrict__ w1, const float* __restrict__ w2,
                         h16* qkvwT, h16* projwT, h16* w1T, h16* w2T) {
  int i = blockIdx.x * 256 + threadIdx.x;
  if (i < 49152) {                       // qkvwT [384][128] = qkv_w[k][n]^T
    int n = i >> 7, k = i & 127;
    qkvwT[i] = (h16)qkv_w[k * 384 + n];
  } else if (i < 65536) {                // projwT [128][128]
    int j = i - 49152; int n = j >> 7, k = j & 127;
    projwT[j] = (h16)proj_w[k * 128 + n];
  } else if (i < 131072) {               // w1T [512][128]
    int j = i - 65536; int n = j >> 7, k = j & 127;
    w1T[j] = (h16)w1[k * 512 + n];
  } else if (i < 196608) {               // w2T [128][512]
    int j = i - 131072; int n = j / 512, k = j % 512;
    w2T[j] = (h16)w2[k * 128 + n];
  }
}

// ---------------- K1/K5: LayerNorm (+ optional shift/window-partition) ----------------
__global__ void k_ln(const float* __restrict__ x, const float* __restrict__ g,
                     const float* __restrict__ be, h16* __restrict__ out, int mode) {
  int wid = blockIdx.x * 8 + (threadIdx.x >> 5);   // one wave per token
  int l = lane_id();
  int src;
  if (mode == 0) {
    int win = wid >> 6, t = wid & 63;
    int b = win >> 6, wrem = win & 63;
    int wi = wrem >> 3, wj = wrem & 7;
    int ti = t >> 3, tj = t & 7;
    int si = ((wi << 3) + ti + SHIFT_) & 63;
    int sj = ((wj << 3) + tj + SHIFT_) & 63;
    src = (b << 12) + (si << 6) + sj;
  } else {
    src = wid;
  }
  v4f v = *(const v4f*)(x + (size_t)src * CC + l * 4);
  float s = v[0] + v[1] + v[2] + v[3];
#pragma unroll
  for (int m = 1; m < 32; m <<= 1) s += __shfl_xor(s, m, 32);
  float mu = s * (1.0f / 128.0f);
  float d0 = v[0] - mu, d1 = v[1] - mu, d2 = v[2] - mu, d3 = v[3] - mu;
  float q = d0 * d0 + d1 * d1 + d2 * d2 + d3 * d3;
#pragma unroll
  for (int m = 1; m < 32; m <<= 1) q += __shfl_xor(q, m, 32);
  float rinv = rsqrtf(q * (1.0f / 128.0f) + 1e-5f);
  v4h o;
#pragma unroll
  for (int i = 0; i < 4; ++i) {
    int c = l * 4 + i;
    float dv = (i == 0 ? d0 : i == 1 ? d1 : i == 2 ? d2 : d3);
    o[i] = (h16)(g[c] * dv * rinv + be[c]);
  }
  *(v4h*)(out + (size_t)wid * CC + l * 4) = o;
}

// ---------------- K2: QKV GEMM per window, scatter into q/k/vT ----------------
// N-tiles are range-uniform: ni = nb + 2j, so j 0..3 -> Q cols, 4..7 -> K, 8..11 -> V.
__global__ void k_qkv(const h16* __restrict__ xw, const h16* __restrict__ wT,
                      const float* __restrict__ bias,
                      h16* __restrict__ qb, h16* __restrict__ kb, h16* __restrict__ vT) {
  int win = blockIdx.x;
  const h16* A = xw + (size_t)win * SS * CC;
  int wv = threadIdx.x >> 5;
  int mi = wv & 3, nb = wv >> 2;
  int l = lane_id();
  int m0 = mi * 16 + ((l >> 4) << 3);

  v16h a[4];
#pragma unroll
  for (int ks = 0; ks < 4; ++ks) a[ks] = load_frag(A + mi * 16 * CC + ks * 32, CC);

  auto store_q = [&](int ni, v8f acc) {
    int n = ni * 16 + (l & 15);
    float bs = bias[n];
    int head = n >> 5, hd = n & 31;
    h16* dst = qb + (((size_t)win * NH + head) * SS + m0) * HD + hd;
#pragma unroll
    for (int r = 0; r < 8; ++r) dst[r * HD] = (h16)((acc[r] + bs) * QSCALE);
  };
  auto store_k = [&](int ni, v8f acc) {
    float bs = bias[ni * 16 + (l & 15)];
    int c = (ni - 8) * 16 + (l & 15);
    int head = c >> 5, hd = c & 31;
    h16* dst = kb + (((size_t)win * NH + head) * SS + m0) * HD + hd;
#pragma unroll
    for (int r = 0; r < 8; ++r) dst[r * HD] = (h16)(acc[r] + bs);
  };
  auto store_v = [&](int ni, v8f acc) {   // transposed: contiguous in m
    float bs = bias[ni * 16 + (l & 15)];
    int c = (ni - 16) * 16 + (l & 15);
    int head = c >> 5, hd = c & 31;
    h16* dst = vT + (((size_t)win * NH + head) * HD + hd) * SS + m0;
#pragma unroll
    for (int r = 0; r < 8; ++r) dst[r] = (h16)(acc[r] + bs);
  };

  v8f t0, t1;
#pragma unroll
  for (int jp = 0; jp < 2; ++jp) {       // Q: ni in {nb, nb+2, nb+4, nb+6}
    int ni0 = nb + 4 * jp;
    gemm4x2(a, wT, ni0, ni0 + 2, &t0, &t1);
    store_q(ni0, t0); store_q(ni0 + 2, t1);
  }
#pragma unroll
  for (int jp = 0; jp < 2; ++jp) {       // K: ni in 8..15
    int ni0 = 8 + nb + 4 * jp;
    gemm4x2(a, wT, ni0, ni0 + 2, &t0, &t1);
    store_k(ni0, t0); store_k(ni0 + 2, t1);
  }
#pragma unroll
  for (int jp = 0; jp < 2; ++jp) {       // V: ni in 16..23
    int ni0 = 16 + nb + 4 * jp;
    gemm4x2(a, wT, ni0, ni0 + 2, &t0, &t1);
    store_v(ni0, t0); store_v(ni0 + 2, t1);
  }
}

// ---------------- K3: windowed attention (1 block/window, 1 wave/head) ----------------
__global__ void k_attn(const h16* __restrict__ qb, const h16* __restrict__ kb,
                       const h16* __restrict__ vT, const float* __restrict__ rpb,
                       h16* __restrict__ ow) {
  __shared__ __align__(16) h16 P[NH][SS * SS];   // 32 KB
  int win = blockIdx.x;
  int h = threadIdx.x >> 5;
  int l = lane_id();
  int hi8 = (l >> 4) << 3;

  const h16* Q  = qb + ((size_t)win * NH + h) * SS * HD;
  const h16* K  = kb + ((size_t)win * NH + h) * SS * HD;
  const h16* VT = vT + ((size_t)win * NH + h) * HD * SS;

  // scores = Q @ K^T   (HD=32 -> single wmma per 16x16 tile)
  v16h aq[4], bk[4];
#pragma unroll
  for (int i = 0; i < 4; ++i) {
    aq[i] = load_frag(Q + i * 16 * HD, HD);
    bk[i] = load_frag(K + i * 16 * HD, HD);
  }
  v8f sc[4][4];
#pragma unroll
  for (int mi = 0; mi < 4; ++mi)
#pragma unroll
    for (int ni = 0; ni < 4; ++ni) {
      v8f z = {};
      sc[mi][ni] = wmma_f16(aq[mi], bk[ni], z);
    }

  // relative-position bias + shifted-window mask (computed arithmetically)
  int wrem = win & 63;
  int wi = wrem >> 3, wj = wrem & 7;
#pragma unroll
  for (int mi = 0; mi < 4; ++mi) {
#pragma unroll
    for (int r = 0; r < 8; ++r) {
      int m = mi * 16 + r + hi8;
      int qi = m >> 3, qj = m & 7;
      int rq = ((wi == 7) ? ((qi < 4) ? 3 : 6) : 0) + ((wj == 7) ? ((qj < 4) ? 1 : 2) : 0);
#pragma unroll
      for (int ni = 0; ni < 4; ++ni) {
        int n = ni * 16 + (l & 15);
        int ki = n >> 3, kj = n & 7;
        int rel = (qi - ki + 7) * 15 + (qj - kj + 7);
        int rk = ((wi == 7) ? ((ki < 4) ? 3 : 6) : 0) + ((wj == 7) ? ((kj < 4) ? 1 : 2) : 0);
        float add = rpb[rel * NH + h] + ((rq != rk) ? -100.0f : 0.0f);
        sc[mi][ni][r] += add;
      }
    }
  }

  // row softmax: row m lives at reg r across 4 ni-tiles in the 16 lanes of one half-wave
#pragma unroll
  for (int mi = 0; mi < 4; ++mi) {
    h16* hp = &P[h][(mi * 16 + hi8) * SS + (l & 15)];
#pragma unroll
    for (int r = 0; r < 8; ++r) {
      float mx = sc[mi][0][r];
#pragma unroll
      for (int ni = 1; ni < 4; ++ni) mx = fmaxf(mx, sc[mi][ni][r]);
#pragma unroll
      for (int m = 1; m < 16; m <<= 1) mx = fmaxf(mx, __shfl_xor(mx, m, 32));
      float sum = 0.0f;
#pragma unroll
      for (int ni = 0; ni < 4; ++ni) {
        float e = __expf(sc[mi][ni][r] - mx);
        sc[mi][ni][r] = e;
        sum += e;
      }
#pragma unroll
      for (int m = 1; m < 16; m <<= 1) sum += __shfl_xor(sum, m, 32);
      float inv = 1.0f / sum;
#pragma unroll
      for (int ni = 0; ni < 4; ++ni)
        hp[r * SS + ni * 16] = (h16)(sc[mi][ni][r] * inv);
    }
  }
  __syncthreads();

  // O = P @ V  (K=64: 2 wmma k-steps of 32; B-frags contiguous from vT)
  v8f oacc[4][2];
#pragma unroll
  for (int mi = 0; mi < 4; ++mi)
#pragma unroll
    for (int ni = 0; ni < 2; ++ni) { v8f z = {}; oacc[mi][ni] = z; }
#pragma unroll
  for (int ki = 0; ki < 2; ++ki) {
    v16h bv0 = load_frag(VT + 0 * 16 * SS + ki * 32, SS);
    v16h bv1 = load_frag(VT + 1 * 16 * SS + ki * 32, SS);
#pragma unroll
    for (int mi = 0; mi < 4; ++mi) {
      v16h ap = load_frag(&P[h][mi * 16 * SS + ki * 32], SS);
      oacc[mi][0] = wmma_f16(ap, bv0, oacc[mi][0]);
      oacc[mi][1] = wmma_f16(ap, bv1, oacc[mi][1]);
    }
  }
  // store O into ow[win][tok][h*32+hd]
#pragma unroll
  for (int mi = 0; mi < 4; ++mi) {
    h16* dst = ow + (((size_t)win * SS) + mi * 16 + hi8) * CC + h * HD + (l & 15);
#pragma unroll
    for (int ni = 0; ni < 2; ++ni)
#pragma unroll
      for (int r = 0; r < 8; ++r)
        dst[r * CC + ni * 16] = (h16)oacc[mi][ni][r];
  }
}

// ---------------- K4: proj GEMM + window-reverse + residual -> x1 (f32) ----------------
__global__ void k_proj(const h16* __restrict__ ow, const h16* __restrict__ wT,
                       const float* __restrict__ pb, const float* __restrict__ x,
                       float* __restrict__ x1) {
  int win = blockIdx.x;
  const h16* A = ow + (size_t)win * SS * CC;
  int wv = threadIdx.x >> 5;
  int mi = wv & 3, nb = wv >> 2;
  int l = lane_id();
  int m0 = mi * 16 + ((l >> 4) << 3);   // multiple of 8 -> tj == r, ti fixed

  int b = win >> 6, wrem = win & 63;
  int wi = wrem >> 3, wj = wrem & 7;
  int ti = m0 >> 3;
  int di = ((wi << 3) + ti + SHIFT_) & 63;
  int rowbase = (b << 12) + (di << 6);
  int wjb = (wj << 3) + SHIFT_;

  v16h a[4];
#pragma unroll
  for (int ks = 0; ks < 4; ++ks) a[ks] = load_frag(A + mi * 16 * CC + ks * 32, CC);

  auto store_o = [&](int ni, v8f acc) {
    int n = ni * 16 + (l & 15);
    float bias = pb[n];
#pragma unroll
    for (int r = 0; r < 8; ++r) {
      int dj = (wjb + r) & 63;
      size_t tok = (size_t)(rowbase + dj);
      x1[tok * CC + n] = x[tok * CC + n] + acc[r] + bias;
    }
  };

  v8f t0, t1;
#pragma unroll
  for (int jp = 0; jp < 2; ++jp) {
    int ni0 = nb + 4 * jp;
    gemm4x2(a, wT, ni0, ni0 + 2, &t0, &t1);
    store_o(ni0, t0); store_o(ni0 + 2, t1);
  }
}

// ---------------- K6: fused MLP (+ residual) -> out ----------------
__global__ void k_mlp(const h16* __restrict__ xn, const h16* __restrict__ w1T,
                      const float* __restrict__ b1, const h16* __restrict__ w2T,
                      const float* __restrict__ b2, const float* __restrict__ x1,
                      float* __restrict__ out) {
  __shared__ __align__(16) h16 Hs[64][256];   // 32 KB half of hidden
  int blk = blockIdx.x;
  size_t row0 = (size_t)blk * 64;
  const h16* A = xn + row0 * CC;
  int wv = threadIdx.x >> 5;
  int mi = wv & 3, nb = wv >> 2;
  int l = lane_id();
  int m0 = mi * 16 + ((l >> 4) << 3);

  v16h a[4];
#pragma unroll
  for (int ks = 0; ks < 4; ++ks) a[ks] = load_frag(A + mi * 16 * CC + ks * 32, CC);

  v8f o[4];
#pragma unroll
  for (int j = 0; j < 4; ++j) { v8f z = {}; o[j] = z; }

  auto store_h = [&](int half, int ni, v8f acc) {
    int n = ni * 16 + (l & 15);
    float bb = b1[n];
    h16* hp = &Hs[m0][n - half * 256];
#pragma unroll
    for (int r = 0; r < 8; ++r) {
      float v = acc[r] + bb;
      hp[r * 256] = (h16)(0.5f * v * (1.0f + erff(v * 0.70710678118654752f)));
    }
  };

#pragma unroll
  for (int half = 0; half < 2; ++half) {
    // phase 1: H[:, half*256 .. +256] = gelu(xn @ W1^T + b1)
    v8f t0, t1;
#pragma unroll
    for (int jp = 0; jp < 4; ++jp) {
      int ni0 = half * 16 + nb + 4 * jp;
      gemm4x2(a, w1T, ni0, ni0 + 2, &t0, &t1);
      store_h(half, ni0, t0); store_h(half, ni0 + 2, t1);
    }
    __syncthreads();
    // phase 2: o += H_half @ W2^T (k-steps half*8 .. half*8+7); 4 independent chains
#pragma unroll 2
    for (int ks = 0; ks < 8; ++ks) {
      int kglob = half * 8 + ks;
      __builtin_prefetch(w2T + nb * 16 * NMLP + (kglob + 1) * 32, 0, 3);
      v16h ah = load_frag(&Hs[mi * 16][ks * 32], 256);
#pragma unroll
      for (int j = 0; j < 4; ++j) {
        int ni = nb + 2 * j;
        v16h bw = load_frag(w2T + ni * 16 * NMLP + kglob * 32, NMLP);
        o[j] = wmma_f16(ah, bw, o[j]);
      }
    }
    __syncthreads();
  }

#pragma unroll
  for (int j = 0; j < 4; ++j) {
    int ni = nb + 2 * j;
    int n = ni * 16 + (l & 15);
    float bb = b2[n];
    size_t base = (row0 + m0) * CC + n;
#pragma unroll
    for (int r = 0; r < 8; ++r)
      out[base + r * CC] = x1[base + r * CC] + o[j][r] + bb;
  }
}

// ---------------- launcher ----------------
extern "C" void kernel_launch(void* const* d_in, const int* in_sizes, int n_in,
                              void* d_out, int out_size, void* d_ws, size_t ws_size,
                              hipStream_t stream) {
  const float* x      = (const float*)d_in[0];
  const float* gamma  = (const float*)d_in[1];
  const float* beta   = (const float*)d_in[2];
  const float* qkv_w  = (const float*)d_in[3];
  const float* qkv_b  = (const float*)d_in[4];
  const float* rpb    = (const float*)d_in[5];
  const float* proj_w = (const float*)d_in[6];
  const float* proj_b = (const float*)d_in[7];
  const float* mlp_w1 = (const float*)d_in[8];
  const float* mlp_b1 = (const float*)d_in[9];
  const float* mlp_w2 = (const float*)d_in[10];
  const float* mlp_b2 = (const float*)d_in[11];
  float* outp = (float*)d_out;

  char* w = (char*)d_ws;
  size_t o = 0;
  h16* qkvwT  = (h16*)(w + o); o += (size_t)384 * 128 * 2;   // 96 KB
  h16* projwT = (h16*)(w + o); o += (size_t)128 * 128 * 2;   // 32 KB
  h16* w1T    = (h16*)(w + o); o += (size_t)512 * 128 * 2;   // 128 KB
  h16* w2T    = (h16*)(w + o); o += (size_t)128 * 512 * 2;   // 128 KB
  const size_t ACT = (size_t)TOK * CC * 2;                   // 32 MB each
  h16* xw   = (h16*)(w + o); o += ACT;                       // LN'd windowed x; reused as xn2
  size_t off_q = o;
  h16* qb   = (h16*)(w + o); o += ACT;
  h16* kb   = (h16*)(w + o); o += ACT;
  h16* vT   = (h16*)(w + o); o += ACT;
  h16* ow   = (h16*)(w + o); o += ACT;
  float* x1 = (float*)(w + off_q);   // alias over q+k (dead after attention), 64 MB f32
  h16* xn2  = xw;                    // alias over xw (dead after QKV)

  k_prep_w<<<768, 256, 0, stream>>>(qkv_w, proj_w, mlp_w1, mlp_w2, qkvwT, projwT, w1T, w2T);
  k_ln<<<TOK / 8, 256, 0, stream>>>(x, gamma, beta, xw, 0);
  k_qkv<<<NWT, 256, 0, stream>>>(xw, qkvwT, qkv_b, qb, kb, vT);
  k_attn<<<NWT, 128, 0, stream>>>(qb, kb, vT, rpb, ow);
  k_proj<<<NWT, 256, 0, stream>>>(ow, projwT, proj_b, x, x1);
  k_ln<<<TOK / 8, 256, 0, stream>>>(x1, gamma, beta, xn2, 1);
  k_mlp<<<TOK / 64, 256, 0, stream>>>(xn2, w1T, mlp_b1, w2T, mlp_b2, x1, outp);
}
